// CausalSelfAttentionRelativePosition_83872121356492
// MI455X (gfx1250) — matrix operand order repaired and checked by reading, not compiled, not run on hardware
//
#include <hip/hip_runtime.h>
#include <stdint.h>

typedef __attribute__((ext_vector_type(16))) __bf16 v16bf;
typedef __attribute__((ext_vector_type(8)))  float  v8f;

#define L_SEQ  2048
#define DMODEL 1024
#define NHEAD  16
#define HS     64
#define BATCH  2
#define N3     3072
#define MROWS  (BATCH * L_SEQ)      // 4096
#define NEG_BIG (-3.0e38f)

union AFrag { uint32_t u[8]; v16bf v; };

static __device__ inline unsigned short f2bf(float f) {
  union { float f; uint32_t u; } x; x.f = f;
  uint32_t u = x.u;
  uint32_t r = (u + 0x7FFFu + ((u >> 16) & 1u)) >> 16;
  return (unsigned short)r;
}

static __device__ inline v8f v8zero() {
  v8f z;
#pragma unroll
  for (int i = 0; i < 8; ++i) z[i] = 0.0f;
  return z;
}

// CDNA5 async DMA: memory -> LDS, no data VGPRs, tracked by ASYNCcnt.
static __device__ inline void async_b128_to_lds(void* lds_ptr, const void* gptr) {
  uint32_t lds = (uint32_t)(uintptr_t)lds_ptr;      // LDS offset = addr[31:0]
  uint64_t ga  = (uint64_t)(uintptr_t)gptr;
  asm volatile("global_load_async_to_lds_b128 %0, %1, off"
               :: "v"(lds), "v"(ga) : "memory");
}
static __device__ inline void wait_async0() {
  asm volatile("s_wait_asynccnt 0x0" ::: "memory");
}

// ---------------------------------------------------------------- f32 -> bf16
__global__ void cvt_bf16_kernel(const float* __restrict__ src,
                                unsigned short* __restrict__ dst, int n) {
  int i = blockIdx.x * blockDim.x + threadIdx.x;
  if (i < n) dst[i] = f2bf(src[i]);
}

// --------------------------------- f32 (K x N) -> bf16 transposed (N x K)
// block (32,8); each block transposes one 32x32 tile
__global__ __launch_bounds__(256) void wtrans_bf16_kernel(
    const float* __restrict__ W, unsigned short* __restrict__ WT,
    int K, int N) {
  __shared__ float tile[32][33];
  int n0 = blockIdx.x * 32;
  int k0 = blockIdx.y * 32;
  int tx = threadIdx.x, ty = threadIdx.y;
#pragma unroll
  for (int j = 0; j < 4; ++j)
    tile[ty + 8 * j][tx] = W[(size_t)(k0 + ty + 8 * j) * N + n0 + tx];
  __syncthreads();
#pragma unroll
  for (int j = 0; j < 4; ++j)
    WT[(size_t)(n0 + ty + 8 * j) * K + k0 + tx] = f2bf(tile[tx][ty + 8 * j]);
}

// ------------------------------------------------- WMMA GEMM (M=4096, K=1024)
// Block tile 128x64 (4 waves, each wave 32x64 -> 8 WMMAs per 32-K step).
// Panels staged with GLOBAL_LOAD_ASYNC_TO_LDS_B128 (ASYNCcnt).
// mode 0: qkv epilogue -> scatter bf16 into head-major q/k (row-major) and
//         v TRANSPOSED (B,H,hs,L).                         (N = 3072)
// mode 1: proj epilogue -> f32 out + bias                  (N = 1024)
__global__ __launch_bounds__(128) void gemm_bf16_kernel(
    const unsigned short* __restrict__ A,   // (4096 x 1024) bf16 row-major
    const unsigned short* __restrict__ BT,  // (N x 1024)    bf16 row-major
    const float* __restrict__ bias,         // (N)
    int N, int mode,
    float* __restrict__ outf,
    unsigned short* __restrict__ qb,
    unsigned short* __restrict__ kb,
    unsigned short* __restrict__ vb) {
  const int K = DMODEL;
  __shared__ uint32_t AsU[128][16];  // 128 rows(m) x 32 bf16 (K chunk)
  __shared__ uint32_t BsU[64][16];   //  64 rows(n) x 32 bf16 (K chunk)

  const int t    = threadIdx.x;
  const int lane = t & 31;
  const int w    = t >> 5;
  const int ln   = lane & 15;
  const int kh   = lane >> 4;
  const int m0b  = blockIdx.x * 128;
  const int n0b  = blockIdx.y * 64;

  const uint4* A4 = (const uint4*)A;
  const uint4* B4 = (const uint4*)BT;
  uint4* As4 = (uint4*)&AsU[0][0];
  uint4* Bs4 = (uint4*)&BsU[0][0];

  v8f acc[2][4];
#pragma unroll
  for (int rt = 0; rt < 2; ++rt)
#pragma unroll
    for (int c = 0; c < 4; ++c) acc[rt][c] = v8zero();

  for (int k0 = 0; k0 < K; k0 += 32) {
    __syncthreads();
    // stage A panel: 128x32 bf16 = 512 uint4, 4 async DMAs per thread
#pragma unroll
    for (int i = 0; i < 4; ++i) {
      int idx = t + 128 * i;            // = row*4 + c4
      int row = idx >> 2, c4 = idx & 3;
      async_b128_to_lds(&As4[idx],
                        &A4[(size_t)(m0b + row) * (K >> 3) + (k0 >> 3) + c4]);
    }
    // stage B panel: 64x32 bf16 = 256 uint4, 2 async DMAs per thread
#pragma unroll
    for (int i = 0; i < 2; ++i) {
      int idx = t + 128 * i;            // = row*4 + c4
      int row = idx >> 2, c4 = idx & 3;
      async_b128_to_lds(&Bs4[idx],
                        &B4[(size_t)(n0b + row) * (K >> 3) + (k0 >> 3) + c4]);
    }
    // prefetch next panels into L2 (global_prefetch_b8)
    if (k0 + 32 < K) {
      __builtin_prefetch((const void*)&A4[(size_t)(m0b + t) * (K >> 3) + ((k0 + 32) >> 3)], 0, 1);
      __builtin_prefetch((const void*)&B4[(size_t)(n0b + (t >> 1)) * (K >> 3) + ((k0 + 32) >> 3)], 0, 1);
    }
    wait_async0();
    __syncthreads();

    AFrag a[2];
#pragma unroll
    for (int rt = 0; rt < 2; ++rt)
#pragma unroll
      for (int v = 0; v < 8; ++v)   // contiguous 8 dwords -> b128 x2
        a[rt].u[v] = AsU[w * 32 + rt * 16 + ln][8 * kh + v];

#pragma unroll
    for (int c = 0; c < 4; ++c) {
      AFrag b;
#pragma unroll
      for (int v = 0; v < 8; ++v)   // contiguous 8 dwords -> b128 x2
        b.u[v] = BsU[c * 16 + ln][8 * kh + v];
      acc[0][c] = __builtin_amdgcn_wmma_f32_16x16x32_bf16(
          false, a[0].v, false, b.v, (short)0, acc[0][c], false, false);
      acc[1][c] = __builtin_amdgcn_wmma_f32_16x16x32_bf16(
          false, a[1].v, false, b.v, (short)0, acc[1][c], false, false);
    }
  }

  // epilogue
#pragma unroll
  for (int rt = 0; rt < 2; ++rt) {
#pragma unroll
    for (int c = 0; c < 4; ++c) {
#pragma unroll
      for (int r = 0; r < 8; ++r) {
        int gm = m0b + w * 32 + rt * 16 + r + 8 * kh;
        int n  = n0b + c * 16 + ln;
        float val = acc[rt][c][r] + bias[n];
        if (mode == 1) {
          outf[(size_t)gm * N + n] = val;
        } else {
          int which = n >> 10;
          int d  = n & 1023;
          int hh = d >> 6, e = d & 63;
          int bb = gm >> 11, ll = gm & (L_SEQ - 1);
          unsigned short bv = f2bf(val);
          if (which == 0) {
            qb[((size_t)((bb * NHEAD + hh) * L_SEQ + ll)) * HS + e] = bv;
          } else if (which == 1) {
            kb[((size_t)((bb * NHEAD + hh) * L_SEQ + ll)) * HS + e] = bv;
          } else {  // v stored transposed: (B,H,hs,L)
            vb[((size_t)((bb * NHEAD + hh) * HS + e)) * L_SEQ + ll] = bv;
          }
        }
      }
    }
  }
}

// ----------------------------------------------------- flash attention + skew
// grid: (L/16, B*H), block: 32 (one wave handles one 16-row q-tile)
// v is (B,H,hs,L) so PV B-fragments are contiguous dwords.
__global__ __launch_bounds__(32) void attn_rel_kernel(
    const unsigned short* __restrict__ qg,
    const unsigned short* __restrict__ kg,
    const unsigned short* __restrict__ vg,   // (B,H,hs,L) bf16
    const unsigned short* __restrict__ er,   // (2048 x 64) bf16
    unsigned short* __restrict__ yb) {       // (B, L, D) bf16
  __shared__ float    Gs[16][48];
  __shared__ uint32_t PsU[16][16];           // 16x32 bf16 (P tile)
  __shared__ uint32_t VsU[64][16];           // V^T tile: 64 rows(hs) x 32 k bf16

  const int lane = threadIdx.x;
  const int ln   = lane & 15;
  const int kh   = lane >> 4;
  const int bh   = blockIdx.y;
  const int q0   = blockIdx.x * 16;

  const uint32_t* qU = (const uint32_t*)(qg + (size_t)bh * L_SEQ * HS);
  const uint32_t* kU = (const uint32_t*)(kg + (size_t)bh * L_SEQ * HS);
  const uint4*    v4 = (const uint4*)(vg + (size_t)bh * HS * L_SEQ);
  const uint32_t* eU = (const uint32_t*)er;
  uint4* Vs4 = (uint4*)&VsU[0][0];

  // Q tile (16x64) as two A operands (hs 0..31, 32..63)
  AFrag aq0, aq1;
  {
    size_t rb = ((size_t)(q0 + ln) * HS) >> 1;
#pragma unroll
    for (int v = 0; v < 8; ++v) {
      int off = 8 * kh + v;
      aq0.u[v] = qU[rb + off];
      aq1.u[v] = qU[rb + 16 + off];
    }
  }

  v8f o[4];
  float m[8], lsum[8];
#pragma unroll
  for (int c = 0; c < 4; ++c) o[c] = v8zero();
#pragma unroll
  for (int r = 0; r < 8; ++r) { m[r] = NEG_BIG; lsum[r] = 0.0f; }

  const float scale = 0.125f;  // 1/sqrt(64)
  const int nk = (q0 + 16 + 31) >> 5;

  for (int kt = 0; kt < nk; ++kt) {
    const int k0 = kt * 32;

    // stage V^T tile (64 rows hs x 32 k bf16) via async DMA, 8 per lane
#pragma unroll
    for (int i = 0; i < 8; ++i) {
      int idx = i * 32 + lane;          // = row*4 + c4
      int row = idx >> 2, c4 = idx & 3;
      async_b128_to_lds(&Vs4[idx],
                        &v4[(size_t)row * (L_SEQ >> 3) + (k0 >> 3) + c4]);
    }
    // prefetch next K rows
    if (k0 + 32 < q0 + 16)
      __builtin_prefetch((const void*)&kU[((size_t)(k0 + 32 + lane) * HS) >> 1], 0, 1);

    // S = Q K^T  (two 16x16 col tiles)
    v8f s[2];
#pragma unroll
    for (int jt = 0; jt < 2; ++jt) {
      AFrag b0, b1;
      size_t rb = ((size_t)(k0 + 16 * jt + ln) * HS) >> 1;
#pragma unroll
      for (int v = 0; v < 8; ++v) {     // contiguous dwords -> b128
        int off = 8 * kh + v;
        b0.u[v] = kU[rb + off];
        b1.u[v] = kU[rb + 16 + off];
      }
      v8f c = v8zero();
      c = __builtin_amdgcn_wmma_f32_16x16x32_bf16(false, aq0.v, false, b0.v,
                                                  (short)0, c, false, false);
      c = __builtin_amdgcn_wmma_f32_16x16x32_bf16(false, aq1.v, false, b1.v,
                                                  (short)0, c, false, false);
      s[jt] = c;
    }

    // relative-position band: G[i,j'] = Q . Er[base0 + j'], j' in [0,48)
    const int base0 = (L_SEQ - 1) - q0 + k0 - 15;
#pragma unroll
    for (int gt = 0; gt < 3; ++gt) {
      AFrag b0, b1;
      int erow = base0 + 16 * gt + ln;
      bool ok = (erow >= 0) && (erow < L_SEQ);
      size_t rb = ok ? (((size_t)erow * HS) >> 1) : 0;
#pragma unroll
      for (int v = 0; v < 8; ++v) {
        int off = 8 * kh + v;
        b0.u[v] = ok ? eU[rb + off] : 0u;
        b1.u[v] = ok ? eU[rb + 16 + off] : 0u;
      }
      v8f g = v8zero();
      g = __builtin_amdgcn_wmma_f32_16x16x32_bf16(false, aq0.v, false, b0.v,
                                                  (short)0, g, false, false);
      g = __builtin_amdgcn_wmma_f32_16x16x32_bf16(false, aq1.v, false, b1.v,
                                                  (short)0, g, false, false);
#pragma unroll
      for (int r = 0; r < 8; ++r)
        Gs[r + 8 * kh][16 * gt + ln] = g[r];
    }
    wait_async0();
    __syncthreads();

    // skewed add, scale, causal mask, online softmax
    unsigned short* Ps16 = (unsigned short*)&PsU[0][0];
#pragma unroll
    for (int r = 0; r < 8; ++r) {
      int i  = r + 8 * kh;
      int gq = q0 + i;
      float s0 = (s[0][r] + Gs[i][15 - i + ln]) * scale;
      float s1 = (s[1][r] + Gs[i][15 - i + 16 + ln]) * scale;
      if (k0 + ln > gq)      s0 = NEG_BIG;
      if (k0 + 16 + ln > gq) s1 = NEG_BIG;
      float mx = fmaxf(s0, s1);
#pragma unroll
      for (int off = 1; off < 16; off <<= 1)
        mx = fmaxf(mx, __shfl_xor(mx, off, 32));
      float mnew = fmaxf(m[r], mx);
      float p0 = __expf(s0 - mnew);
      float p1 = __expf(s1 - mnew);
      float rs = p0 + p1;
#pragma unroll
      for (int off = 1; off < 16; off <<= 1)
        rs += __shfl_xor(rs, off, 32);
      float corr = __expf(m[r] - mnew);
      lsum[r] = lsum[r] * corr + rs;
      m[r] = mnew;
#pragma unroll
      for (int c = 0; c < 4; ++c) o[c][r] *= corr;
      Ps16[i * 32 + ln]      = f2bf(p0);
      Ps16[i * 32 + 16 + ln] = f2bf(p1);
    }
    __syncthreads();

    // O += P V   (P: 16x32 A operand from LDS; V^T: contiguous B fragments)
    AFrag ap;
#pragma unroll
    for (int v = 0; v < 8; ++v)       // contiguous 8 dwords -> b128 x2
      ap.u[v] = PsU[ln][8 * kh + v];
#pragma unroll
    for (int c = 0; c < 4; ++c) {
      AFrag bv;
#pragma unroll
      for (int v = 0; v < 8; ++v)     // contiguous 8 dwords -> b128 x2
        bv.u[v] = VsU[c * 16 + ln][8 * kh + v];
      o[c] = __builtin_amdgcn_wmma_f32_16x16x32_bf16(
          false, ap.v, false, bv.v, (short)0, o[c], false, false);
    }
    __syncthreads();
  }

  // epilogue: y[b, q, h*64+e] = O / l   (bf16)
  const int bb = bh >> 4, hh = bh & 15;
#pragma unroll
  for (int r = 0; r < 8; ++r) {
    float rl = 1.0f / lsum[r];
    int row = q0 + r + 8 * kh;
    size_t base = ((size_t)(bb * L_SEQ + row)) * DMODEL + hh * HS;
#pragma unroll
    for (int c = 0; c < 4; ++c)
      yb[base + 16 * c + ln] = f2bf(o[c][r] * rl);
  }
}

// ---------------------------------------------------------------------- host
extern "C" void kernel_launch(void* const* d_in, const int* in_sizes, int n_in,
                              void* d_out, int out_size, void* d_ws, size_t ws_size,
                              hipStream_t stream) {
  const float* x      = (const float*)d_in[0];
  const float* W_attn = (const float*)d_in[1];
  const float* b_attn = (const float*)d_in[2];
  const float* W_proj = (const float*)d_in[3];
  const float* b_proj = (const float*)d_in[4];
  const float* Er     = (const float*)d_in[5];

  char* ws = (char*)d_ws;
  size_t off = 0;
  unsigned short* xb  = (unsigned short*)(ws + off); off += (size_t)MROWS * DMODEL * 2;
  unsigned short* wab = (unsigned short*)(ws + off); off += (size_t)DMODEL * N3 * 2;     // transposed (N3 x K)
  unsigned short* wpb = (unsigned short*)(ws + off); off += (size_t)DMODEL * DMODEL * 2; // transposed (N x K)
  unsigned short* erb = (unsigned short*)(ws + off); off += (size_t)L_SEQ * HS * 2;
  unsigned short* qb  = (unsigned short*)(ws + off); off += (size_t)MROWS * DMODEL * 2;
  unsigned short* kb  = (unsigned short*)(ws + off); off += (size_t)MROWS * DMODEL * 2;
  unsigned short* vb  = (unsigned short*)(ws + off); off += (size_t)MROWS * DMODEL * 2;  // transposed (B,H,hs,L)
  unsigned short* yb  = (unsigned short*)(ws + off); off += (size_t)MROWS * DMODEL * 2;

  const int nx  = MROWS * DMODEL;          // 4194304
  const int ner = L_SEQ * HS;              // 131072
  cvt_bf16_kernel<<<(nx  + 255) / 256, 256, 0, stream>>>(x,  xb,  nx);
  cvt_bf16_kernel<<<(ner + 255) / 256, 256, 0, stream>>>(Er, erb, ner);
  wtrans_bf16_kernel<<<dim3(N3 / 32, DMODEL / 32), dim3(32, 8), 0, stream>>>(
      W_attn, wab, DMODEL, N3);
  wtrans_bf16_kernel<<<dim3(DMODEL / 32, DMODEL / 32), dim3(32, 8), 0, stream>>>(
      W_proj, wpb, DMODEL, DMODEL);

  gemm_bf16_kernel<<<dim3(MROWS / 128, N3 / 64), 128, 0, stream>>>(
      xb, wab, b_attn, N3, 0, nullptr, qb, kb, vb);

  attn_rel_kernel<<<dim3(L_SEQ / 16, BATCH * NHEAD), 32, 0, stream>>>(
      qb, kb, vb, erb, yb);

  gemm_bf16_kernel<<<dim3(MROWS / 128, DMODEL / 64), 128, 0, stream>>>(
      yb, wpb, b_proj, DMODEL, 1, (float*)d_out, nullptr, nullptr, nullptr);
}
